// VectorQuantizer_16028817948696
// MI455X (gfx1250) — matrix operand order repaired and checked by reading, not compiled, not run on hardware
//
#include <hip/hip_runtime.h>
#include <hip/hip_bf16.h>
#include <cstdint>
#include <cstddef>

// ---------------------------------------------------------------------------
// VQ-VAE vector quantizer for MI455X (gfx1250, wave32, WMMA).
// distances = ||f||^2 + ||c||^2 - 2 f.c ; argmin over ||c||^2 - 2 f.c (f-term
// is row-constant). GEMM via v_wmma_f32_16x16x32_bf16; codebook tiles streamed
// into double-buffered LDS with global_load_async_to_lds_b128 (ASYNCcnt).
// ---------------------------------------------------------------------------

typedef __attribute__((ext_vector_type(16))) __bf16 v16bf;
typedef __attribute__((ext_vector_type(8)))  __bf16 v8bf;
typedef __attribute__((ext_vector_type(8)))  float  v8f;

#define VQ_D      512
#define VQ_K      8192
#define VQ_N      32768
#define NTILES    (VQ_K / 16)
#define LDS_PITCH 520   // 512 + 8 ushort pad -> +16B per row, breaks bank conflicts

__device__ __forceinline__ unsigned short f2bf_u(float f) {
  unsigned int u = __builtin_bit_cast(unsigned int, f);
  unsigned int lsb = (u >> 16) & 1u;
  u += 0x7fffu + lsb;                 // round to nearest even
  return (unsigned short)(u >> 16);
}
__device__ __forceinline__ __bf16 f2bf(float f) {
  return __builtin_bit_cast(__bf16, f2bf_u(f));
}

__device__ __forceinline__ void wait_asynccnt0() {
#if __has_builtin(__builtin_amdgcn_s_wait_asynccnt)
  __builtin_amdgcn_s_wait_asynccnt(0);
#else
  asm volatile("s_wait_asynccnt 0x0" ::: "memory");
#endif
}

// Issue 64 contiguous bytes: global (saddr + vaddr) -> LDS[lds_addr], async.
// Per ISA 15.18: INST_OFFSET is added to BOTH the global and the LDS address.
__device__ __forceinline__ void async_copy64(unsigned int lds_addr,
                                             unsigned int vaddr,
                                             const void* saddr) {
  asm volatile(
      "global_load_async_to_lds_b128 %0, %1, %2\n"
      "global_load_async_to_lds_b128 %0, %1, %2 offset:16\n"
      "global_load_async_to_lds_b128 %0, %1, %2 offset:32\n"
      "global_load_async_to_lds_b128 %0, %1, %2 offset:48\n"
      :
      : "v"(lds_addr), "v"(vaddr), "s"(saddr)
      : "memory");
}

// --- Kernel 1: codebook fp32 -> bf16, plus ||c||^2 ------------------------
__global__ __launch_bounds__(256) void vq_prep(const float* __restrict__ cb,
                                               unsigned short* __restrict__ cbh,
                                               float* __restrict__ norms) {
  __shared__ float red[256];
  const int row = blockIdx.x;
  const int tid = threadIdx.x;
  const float* src = cb + (size_t)row * VQ_D;
  float2 v = *(const float2*)(src + tid * 2);
  unsigned int packed = (unsigned int)f2bf_u(v.x) | ((unsigned int)f2bf_u(v.y) << 16);
  ((unsigned int*)cbh)[(size_t)row * (VQ_D / 2) + tid] = packed;
  red[tid] = v.x * v.x + v.y * v.y;
  __syncthreads();
  for (int off = 128; off > 0; off >>= 1) {
    if (tid < off) red[tid] += red[tid + off];
    __syncthreads();
  }
  if (tid == 0) norms[row] = red[0];
}

// --- Kernel 2: WMMA argmin ------------------------------------------------
// Block = 256 threads = 8 waves; wave w owns row-strip (blockIdx*8 + w)*16.
// A (16 rows x 512) held in registers as bf16; codebook tiles (16 codes x 512)
// double-buffered in LDS via async loads, shared by all 8 waves.
__global__ __launch_bounds__(256, 1) void vq_argmin(const float* __restrict__ x,
                                                    const unsigned short* __restrict__ cbh,
                                                    const float* __restrict__ norms,
                                                    int* __restrict__ indices) {
  __shared__ __align__(16) unsigned short tile[2][16 * LDS_PITCH];

  const int tid  = threadIdx.x;
  const int wave = tid >> 5;
  const int lane = tid & 31;
  const int col  = lane & 15;     // B column / C column for this lane
  const int sel  = lane >> 4;     // which K-half this lane holds
  const int strip = blockIdx.x * 8 + wave;
  const int row   = strip * 16 + col;   // A row held by this lane

  // This thread's 64-byte staging slice of each tile.
  const int tidrow = tid >> 4;          // 0..15 : code row this thread copies
  const int tidoff = (tid & 15) * 32;   // 32 contiguous ushorts per thread
  const unsigned int lds_slice[2] = {
      (unsigned int)(unsigned long long)&tile[0][tidrow * LDS_PITCH + tidoff],
      (unsigned int)(unsigned long long)&tile[1][tidrow * LDS_PITCH + tidoff]};

  // Prologue: kick off tile 0 while we load/convert A below.
  {
    unsigned int vaddr = (unsigned int)(((0 * 16 + tidrow) * VQ_D + tidoff) * 2);
    async_copy64(lds_slice[0], vaddr, cbh);
  }

  // Load this wave's 16 x-rows into WMMA A layout (ISA 7.12.2, 16-bit A 16x32):
  // lanes 0-15: K = {c*32+0..7, c*32+16..23}; lanes 16-31: +8 on each group.
  v16bf a[16];
  {
    const float* xr = x + (size_t)row * VQ_D;
    #pragma unroll
    for (int c = 0; c < 16; ++c) {
      const int b0 = c * 32 + sel * 8;
      const int b1 = c * 32 + 16 + sel * 8;
      float4 f0 = *(const float4*)(xr + b0);
      float4 f1 = *(const float4*)(xr + b0 + 4);
      float4 f2 = *(const float4*)(xr + b1);
      float4 f3 = *(const float4*)(xr + b1 + 4);
      a[c][0]  = f2bf(f0.x); a[c][1]  = f2bf(f0.y); a[c][2]  = f2bf(f0.z); a[c][3]  = f2bf(f0.w);
      a[c][4]  = f2bf(f1.x); a[c][5]  = f2bf(f1.y); a[c][6]  = f2bf(f1.z); a[c][7]  = f2bf(f1.w);
      a[c][8]  = f2bf(f2.x); a[c][9]  = f2bf(f2.y); a[c][10] = f2bf(f2.z); a[c][11] = f2bf(f2.w);
      a[c][12] = f2bf(f3.x); a[c][13] = f2bf(f3.y); a[c][14] = f2bf(f3.z); a[c][15] = f2bf(f3.w);
    }
  }

  float bestv[8];
  int   besti[8];
  #pragma unroll
  for (int r = 0; r < 8; ++r) { bestv[r] = 3.4e38f; besti[r] = 0; }

  for (int t = 0; t < NTILES; ++t) {
    const int buf = t & 1;
    // My async loads for tile t (issued last iteration / prologue) complete;
    // barrier publishes every wave's slice and closes out iter t-1 compute.
    wait_asynccnt0();
    __syncthreads();
    // Prefetch tile t+1 into the other buffer (safe: all waves finished
    // reading it -- they computed tile t-1 from it before the barrier above).
    if (t + 1 < NTILES) {
      unsigned int vaddr = (unsigned int)((((t + 1) * 16 + tidrow) * VQ_D + tidoff) * 2);
      async_copy64(lds_slice[buf ^ 1], vaddr, cbh);
    }

    v8f acc = {};   // accumulates f . c over D = 512
    #pragma unroll
    for (int c = 0; c < 16; ++c) {
      // B layout (32x16 bf16): lane holds column `col`, K = {sel*8+0..7, 16+sel*8+0..7}.
      const v8bf* bp = (const v8bf*)(&tile[buf][col * LDS_PITCH + c * 32 + sel * 8]);
      v8bf lo = bp[0], hi = bp[2];
      v16bf b;
      #pragma unroll
      for (int i = 0; i < 8; ++i) { b[i] = lo[i]; b[i + 8] = hi[i]; }
      acc = __builtin_amdgcn_wmma_f32_16x16x32_bf16(
          /*neg_a=*/false, a[c], /*neg_b=*/false, b,
          /*c_mod=*/(short)0, acc, /*reuse_a=*/false, /*reuse_b=*/false);
    }

    const int code = t * 16 + col;
    const float nrm = norms[code];
    #pragma unroll
    for (int r = 0; r < 8; ++r) {
      float s = fmaf(-2.0f, acc[r], nrm);          // ||c||^2 - 2 f.c
      if (s < bestv[r]) { bestv[r] = s; besti[r] = code; }
    }
  }

  // Min-reduce across the 16 lanes that share each C row (masks 1..8 stay
  // inside each 16-lane half of the wave). Tie-break: lowest code index.
  #pragma unroll
  for (int r = 0; r < 8; ++r) {
    float v = bestv[r]; int i = besti[r];
    #pragma unroll
    for (int m = 8; m >= 1; m >>= 1) {
      float ov = __shfl_xor(v, m, 32);
      int   oi = __shfl_xor(i, m, 32);
      if (ov < v || (ov == v && oi < i)) { v = ov; i = oi; }
    }
    bestv[r] = v; besti[r] = i;
  }
  if (col == 0) {                       // lane 0 -> rows 0..7, lane 16 -> rows 8..15
    const int base = strip * 16 + sel * 8;
    #pragma unroll
    for (int r = 0; r < 8; ++r) indices[base + r] = besti[r];
  }
}

// --- Kernel 3: gather + straight-through output + per-row loss partial ----
__global__ __launch_bounds__(256) void vq_gather(const float* __restrict__ x,
                                                 const float* __restrict__ cb,
                                                 const int* __restrict__ indices,
                                                 float* __restrict__ out,
                                                 float* __restrict__ partial) {
  __shared__ float red[256];
  const int row = blockIdx.x;
  const int tid = threadIdx.x;
  const int idx = indices[row];
  const float* q  = cb + (size_t)idx * VQ_D;
  const float* xr = x  + (size_t)row * VQ_D;
  float* o = out + (size_t)row * VQ_D;
  float s = 0.f;
  #pragma unroll
  for (int j = 0; j < 2; ++j) {
    const int d = tid + j * 256;
    float xv = xr[d];
    float qv = q[d];
    float diff = qv - xv;
    o[d] = xv + diff;                   // x + stop_grad(q - x)
    s += diff * diff;
  }
  red[tid] = s;
  __syncthreads();
  for (int off = 128; off > 0; off >>= 1) {
    if (tid < off) red[tid] += red[tid + off];
    __syncthreads();
  }
  if (tid == 0) partial[row] = red[0];
}

// --- Kernel 4: deterministic final loss reduction -------------------------
__global__ __launch_bounds__(256) void vq_loss(const float* __restrict__ partial,
                                               float* __restrict__ loss) {
  __shared__ float red[256];
  float s = 0.f;
  for (int j = threadIdx.x; j < VQ_N; j += 256) s += partial[j];
  red[threadIdx.x] = s;
  __syncthreads();
  for (int off = 128; off > 0; off >>= 1) {
    if (threadIdx.x < off) red[threadIdx.x] += red[threadIdx.x + off];
    __syncthreads();
  }
  if (threadIdx.x == 0) loss[0] = 0.25f * red[0] / (float)(VQ_N * VQ_D);
}

// ---------------------------------------------------------------------------
extern "C" void kernel_launch(void* const* d_in, const int* in_sizes, int n_in,
                              void* d_out, int out_size, void* d_ws, size_t ws_size,
                              hipStream_t stream) {
  const float* x  = (const float*)d_in[0];   // [32,1024,512] fp32
  const float* cb = (const float*)d_in[1];   // [8192,512]    fp32

  char* ws = (char*)d_ws;
  unsigned short* cbh   = (unsigned short*)ws;                                   // 8 MB bf16 codebook
  size_t off = (size_t)VQ_K * VQ_D * sizeof(unsigned short);
  float* norms          = (float*)(ws + off);  off += (size_t)VQ_K * sizeof(float);   // 32 KB
  int*   indices        = (int*)  (ws + off);  off += (size_t)VQ_N * sizeof(int);     // 128 KB
  float* partial        = (float*)(ws + off);                                          // 128 KB

  float* out  = (float*)d_out;               // quantized_st, 16,777,216 floats
  float* loss = out + (size_t)VQ_N * VQ_D;   // + scalar commitment loss

  vq_prep  <<<VQ_K,          256, 0, stream>>>(cb, cbh, norms);
  vq_argmin<<<VQ_N / 16 / 8, 256, 0, stream>>>(x, cbh, norms, indices);
  vq_gather<<<VQ_N,          256, 0, stream>>>(x, cb, indices, out, partial);
  vq_loss  <<<1,             256, 0, stream>>>(partial, loss);
}
